// KVMN_14139032338670
// MI455X (gfx1250) — matrix operand order repaired and checked by reading, not compiled, not run on hardware
//
#include <hip/hip_runtime.h>
#include <hip/hip_bf16.h>

#define B_   4
#define S_   512
#define M_   128
#define H_   768

typedef __attribute__((ext_vector_type(16))) __bf16 v16bf;
typedef __attribute__((ext_vector_type(8)))  float  v8f;

// Round-to-nearest-even f32->bf16 for a pair, packed into one dword with
// v_perm_b32: result[15:0] = hi16(rne(lo)), result[31:16] = hi16(rne(hi)).
static __device__ __forceinline__ unsigned pack_bf16x2(float lo, float hi) {
    union { float f; unsigned u; } a, b;
    a.f = lo; b.f = hi;
    unsigned ua = a.u + 0x7FFFu + ((a.u >> 16) & 1u);
    unsigned ub = b.u + 0x7FFFu + ((b.u >> 16) & 1u);
    // S0 = ub, S1 = ua; sel bytes {7,6,3,2} -> {ub[3],ub[2],ua[3],ua[2]}
    return __builtin_amdgcn_perm(ub, ua, 0x07060302u);
}

__global__ __launch_bounds__(256) void kvmn_fused(
    const int*   __restrict__ word_seq,     // [B, M]
    const float* __restrict__ hidden,       // [B, S, H]
    const int*   __restrict__ lvm,          // [B, S, M]
    const float* __restrict__ mask,         // [B, S, M]
    const float* __restrict__ key_table,    // [KEY_SIZE, H]
    const float* __restrict__ value_table,  // [VALUE_SIZE, H]
    float*       __restrict__ out)          // [B, S, H]
{
    __shared__ float pbuf[16 * M_];     // exp(u)*mask, then normalized p
    __shared__ int   idxbuf[16 * M_];   // gathered value indices for this s-tile
    __shared__ float rowinv[16];

    const int tiles_per_b = S_ / 16;                // 32
    const int b  = blockIdx.x / tiles_per_b;
    const int s0 = (blockIdx.x % tiles_per_b) * 16;

    const int tid  = threadIdx.x;                   // 0..255
    const int wave = tid >> 5;                      // 0..7 -> m-column tile
    const int lane = tid & 31;
    const int lrow = lane & 15;
    const int lgrp = lane >> 4;

    // ------------------------------------------------------------------
    // Phase 1: u[s0..s0+15, 16w..16w+15] = hidden x key_table[word_seq]^T
    // via v_wmma_f32_16x16x32_bf16, K swept over H in steps of 32.
    // ------------------------------------------------------------------
    const int   mcol = wave * 16 + lrow;                        // B-matrix column (m)
    const int   kidx = word_seq[b * M_ + mcol];                 // gathered key row
    const float* arow = hidden    + ((size_t)b * S_ + (s0 + lrow)) * H_;
    const float* brow = key_table + (size_t)kidx * H_;

    v8f acc = {};
    #pragma unroll 4
    for (int kk = 0; kk < H_; kk += 32) {
        // A operand: 16-bit A 16x32 layout. Lane group g = l>>4:
        //   dwords 0..3  = K (kk + 8g)      .. (kk + 8g + 7)
        //   dwords 4..7  = K (kk + 16 + 8g) .. (kk + 16 + 8g + 7)
        union { v16bf v; unsigned d[8]; } A;
        const float* a0 = arow + kk +      8 * lgrp;
        const float* a1 = arow + kk + 16 + 8 * lgrp;
        #pragma unroll
        for (int i = 0; i < 4; ++i) {
            A.d[i]     = pack_bf16x2(a0[2 * i], a0[2 * i + 1]);
            A.d[i + 4] = pack_bf16x2(a1[2 * i], a1[2 * i + 1]);
        }
        // B operand: 16-bit B 32x16 layout. Lane = column (mcol),
        // same K order as A: K (kk + 8g)..+7, then K (kk + 16 + 8g)..+7.
        union { v16bf v; unsigned d[8]; } Bv;
        const float* b0 = brow + kk +      8 * lgrp;
        const float* b1 = brow + kk + 16 + 8 * lgrp;
        #pragma unroll
        for (int i = 0; i < 4; ++i) {
            Bv.d[i]     = pack_bf16x2(b0[2 * i], b0[2 * i + 1]);
            Bv.d[i + 4] = pack_bf16x2(b1[2 * i], b1[2 * i + 1]);
        }

        acc = __builtin_amdgcn_wmma_f32_16x16x32_bf16(
                  false, A.v, false, Bv.v, (short)0, acc, false, false);
    }

    // ------------------------------------------------------------------
    // Scatter exp(u/sqrt(H)) * clip(mask,0,1) into LDS per C/D layout:
    // acc element r -> row (r + 8*lgrp), column mcol of this tile.
    // ------------------------------------------------------------------
    const float inv_temper = 0.03608439182435161f;  // 1/sqrt(768)
    #pragma unroll
    for (int r = 0; r < 8; ++r) {
        const int srow = r + 8 * lgrp;
        float mk = mask[((size_t)b * S_ + (s0 + srow)) * M_ + mcol];
        mk = fminf(fmaxf(mk, 0.0f), 1.0f);
        pbuf[srow * M_ + mcol] = __expf(acc[r] * inv_temper) * mk;
    }

    // Stage value-gather indices for the whole tile (8 per thread).
    #pragma unroll
    for (int j = 0; j < 8; ++j) {
        const int i    = tid + 256 * j;             // 0..2047
        const int srow = i >> 7;
        const int m    = i & (M_ - 1);
        idxbuf[i] = lvm[((size_t)b * S_ + (s0 + srow)) * M_ + m];
    }
    __syncthreads();

    // Row sums -> reciprocal; then normalize p in LDS.
    if (tid < 16) {
        float s = 1e-10f;
        #pragma unroll 4
        for (int m = 0; m < M_; ++m) s += pbuf[tid * M_ + m];
        rowinv[tid] = 1.0f / s;
    }
    __syncthreads();
    #pragma unroll
    for (int j = 0; j < 8; ++j) {
        const int i = tid + 256 * j;
        pbuf[i] *= rowinv[i >> 7];
    }
    __syncthreads();

    // ------------------------------------------------------------------
    // Phase 2: o[s, :] = sum_m p[s,m] * value_table[idx[s,m], :]
    // 256 threads x 3 h-columns each = 768; coalesced 1KB row reads
    // (value_table is L2-resident) with prefetch of the next row.
    // ------------------------------------------------------------------
    const int h0 = tid, h1 = tid + 256, h2 = tid + 512;
    for (int s = 0; s < 16; ++s) {
        const float* pb = pbuf   + s * M_;
        const int*   ib = idxbuf + s * M_;
        float f0 = 0.0f, f1 = 0.0f, f2 = 0.0f;
        const float* vr = value_table + (size_t)ib[0] * H_;
        #pragma unroll 2
        for (int m = 0; m < M_; ++m) {
            const float pm = pb[m];
            if (m + 1 < M_) {
                const float* vn = value_table + (size_t)ib[m + 1] * H_;
                __builtin_prefetch(vn + h0, 0, 0);   // global_prefetch_b8
                f0 = fmaf(pm, vr[h0], f0);
                f1 = fmaf(pm, vr[h1], f1);
                f2 = fmaf(pm, vr[h2], f2);
                vr = vn;
            } else {
                f0 = fmaf(pm, vr[h0], f0);
                f1 = fmaf(pm, vr[h1], f1);
                f2 = fmaf(pm, vr[h2], f2);
            }
        }
        const size_t ob = ((size_t)b * S_ + (s0 + s)) * H_;
        out[ob + h0] = f0;
        out[ob + h1] = f1;
        out[ob + h2] = f2;
    }
}

extern "C" void kernel_launch(void* const* d_in, const int* in_sizes, int n_in,
                              void* d_out, int out_size, void* d_ws, size_t ws_size,
                              hipStream_t stream) {
    const int*   word_seq = (const int*)  d_in[0];
    const float* hidden   = (const float*)d_in[1];
    const int*   lvm      = (const int*)  d_in[2];
    const float* mask     = (const float*)d_in[3];
    const float* keyt     = (const float*)d_in[4];
    const float* valt     = (const float*)d_in[5];
    float* out = (float*)d_out;

    (void)in_sizes; (void)n_in; (void)out_size; (void)d_ws; (void)ws_size;

    dim3 grid(B_ * (S_ / 16));   // 128 workgroups, 8 waves each
    kvmn_fused<<<grid, 256, 0, stream>>>(word_seq, hidden, lvm, mask, keyt, valt, out);
}
